// GCN_alignment_cnn_47502338294133
// MI455X (gfx1250) — matrix operand order repaired and checked by reading, not compiled, not run on hardware
//
#include <hip/hip_runtime.h>

#define NN 8192
#define HH 256
#define BB 32
#define LL 128
#define PADH 160
#define PADW 544

#define SCHED_FENCE() __builtin_amdgcn_sched_barrier(0)

typedef __attribute__((ext_vector_type(16))) __bf16 v16bf;
typedef __attribute__((ext_vector_type(8)))  float  v8f;

union ABf {
  v16bf v;
  uint4 q[2];
  unsigned int u[8];
  unsigned short s[16];
};

__device__ __forceinline__ unsigned short f2bf(float f) {
  unsigned int u = __builtin_bit_cast(unsigned int, f);
  u += 0x7FFFu + ((u >> 16) & 1u);   // round-to-nearest-even
  return (unsigned short)(u >> 16);
}
__device__ __forceinline__ float bf2f(unsigned short h) {
  unsigned int u = ((unsigned int)h) << 16;
  return __builtin_bit_cast(float, u);
}
// pack two floats -> two bf16 (round-half-up) with one v_perm_b32
__device__ __forceinline__ unsigned int pk2bf(float x, float y) {
  unsigned int ux = __builtin_bit_cast(unsigned int, x) + 0x8000u;
  unsigned int uy = __builtin_bit_cast(unsigned int, y) + 0x8000u;
  return __builtin_amdgcn_perm(uy, ux, 0x07060302u);
}

__device__ __forceinline__ v8f wmma_bf16(const ABf& a, const ABf& b, v8f c) {
  return __builtin_amdgcn_wmma_f32_16x16x32_bf16(false, a.v, false, b.v,
                                                 (short)0, c, false, false);
}

// 32B (2 x b128) global -> LDS async copy, GVS addressing
__device__ __forceinline__ void async_row32(unsigned lds, unsigned goff, const void* sbase) {
  asm volatile(
      "global_load_async_to_lds_b128 %0, %1, %2\n\t"
      "global_load_async_to_lds_b128 %0, %1, %2 offset:16"
      :: "v"(lds), "v"(goff), "s"(sbase) : "memory");
}

// ---------------------------------------------------------------- utilities
__global__ void k_zero(unsigned int* __restrict__ p, long long n) {
  long long i = (long long)blockIdx.x * 256 + threadIdx.x;
  long long stride = (long long)gridDim.x * 256;
  for (; i < n; i += stride) p[i] = 0u;
}

// embT[h][n] = bf16(emb[n][h])   (B-operand for layer1 GEMM: contiguous rows)
__global__ void k_embT(const float* __restrict__ emb, unsigned short* __restrict__ embT) {
  int i = blockIdx.x * 256 + threadIdx.x;
  if (i >= NN * HH) return;
  int n = i >> 8;
  int h = i & 255;
  embT[(size_t)h * NN + n] = f2bf(emb[i]);
}

// ------------------------------------------------- layer1 = adj @ embedding
// grid (64, 2) x 512 threads (16 waves): 128-row M-strip, 2-way K split.
// BOTH operand slabs (adj 128x32 fp32 + embT 256x32 bf16, 32KB/step) stream
// through async global->LDS (ASYNCcnt), double buffered -> no LOADcnt in the
// hot loop; the pre-burst wait is dscnt-only. 8-WMMA bursts per wave.
__global__ __launch_bounds__(512) void k_gemm1(const float* __restrict__ adj,
                                               const unsigned short* __restrict__ embT,
                                               float* __restrict__ part) {
  __shared__ __align__(16) unsigned short Bs[2][256][40];  // 80B row stride
  __shared__ __align__(16) float As[2][128][36];           // 144B row stride
  const int tid  = threadIdx.x;
  const int wave = tid >> 5;
  const int lane = tid & 31;
  const int lrow = lane & 15;
  const int hi   = (lane >> 4) & 1;
  const int wrow = wave >> 1;     // 0..7  -> M sub-strip
  const int wcol = wave & 1;      // 0..1  -> N half (tiles 8*wcol..)
  const int m0   = blockIdx.x * 128 + wrow * 16;
  const int split = blockIdx.y;

  // B staging: thread -> (row = tid/2, 32B half = tid&1)
  const int sBrow = tid >> 1, sBpart = tid & 1;
  const unsigned ldsB0 = (unsigned)(uintptr_t)(&Bs[0][sBrow][sBpart * 16]);
  const unsigned ldsB1 = (unsigned)(uintptr_t)(&Bs[1][sBrow][sBpart * 16]);
  // A staging: thread -> (row = tid/4, 32B quarter = tid&3)
  const int sArow = tid >> 2, sApart = tid & 3;
  const unsigned ldsA0 = (unsigned)(uintptr_t)(&As[0][sArow][sApart * 8]);
  const unsigned ldsA1 = (unsigned)(uintptr_t)(&As[1][sArow][sApart * 8]);
  const size_t gArow = (size_t)(blockIdx.x * 128 + sArow) * NN;

  v8f acc[8] = {};
  const int k0 = split * (NN / 2);
  const int NS = (NN / 2) / 32;

  async_row32(ldsA0, (unsigned)((gArow + k0) * 4 + sApart * 32), adj);
  async_row32(ldsB0, (unsigned)(((size_t)sBrow * NN + k0) * 2 + sBpart * 32), embT);

  for (int s = 0; s < NS; ++s) {
    const int kk = k0 + s * 32;
    if (s + 1 < NS) {
      const unsigned dA = ((s + 1) & 1) ? ldsA1 : ldsA0;
      const unsigned dB = ((s + 1) & 1) ? ldsB1 : ldsB0;
      async_row32(dA, (unsigned)((gArow + kk + 32) * 4 + sApart * 32), adj);
      async_row32(dB, (unsigned)(((size_t)sBrow * NN + kk + 32) * 2 + sBpart * 32), embT);
      asm volatile("s_wait_asynccnt 0x4" ::: "memory");  // current slabs landed
    } else {
      asm volatile("s_wait_asynccnt 0x0" ::: "memory");
    }
    __syncthreads();

    // one DS clause: 4 A-loads first, then 16 B-loads (distinct VGPRs)
    const float* Ap = &As[s & 1][wrow * 16 + lrow][0];
    float4 g0 = *(const float4*)(Ap + hi * 8);
    float4 g1 = *(const float4*)(Ap + hi * 8 + 4);
    float4 g2 = *(const float4*)(Ap + 16 + hi * 8);
    float4 g3 = *(const float4*)(Ap + 16 + hi * 8 + 4);
    const unsigned short (*Bp)[40] = Bs[s & 1];
    ABf bb[8];
#pragma unroll
    for (int t = 0; t < 8; ++t) {
      const uint4* bp = (const uint4*)(&Bp[(wcol * 8 + t) * 16 + lrow][hi * 8]);
      bb[t].q[0] = bp[0];
      bb[t].q[1] = bp[2];
    }
    SCHED_FENCE();

    // pack A (waits only for the 4 A ds-loads; B drains underneath)
    ABf a;
    a.u[0] = pk2bf(g0.x, g0.y);
    a.u[1] = pk2bf(g0.z, g0.w);
    a.u[2] = pk2bf(g1.x, g1.y);
    a.u[3] = pk2bf(g1.z, g1.w);
    a.u[4] = pk2bf(g2.x, g2.y);
    a.u[5] = pk2bf(g2.z, g2.w);
    a.u[6] = pk2bf(g3.x, g3.y);
    a.u[7] = pk2bf(g3.z, g3.w);
    SCHED_FENCE();

#pragma unroll
    for (int t = 0; t < 8; ++t)
      acc[t] = wmma_bf16(a, bb[t], acc[t]);
    __syncthreads();  // slabs may be overwritten two steps ahead
  }

  float* out = part + (size_t)split * NN * HH;
#pragma unroll
  for (int t = 0; t < 8; ++t)
#pragma unroll
    for (int r = 0; r < 8; ++r)
      out[(size_t)(m0 + r + hi * 8) * HH + (wcol * 8 + t) * 16 + lrow] = acc[t][r];
}

__global__ void k_add(const float* __restrict__ p, float* __restrict__ layer1) {
  int i = blockIdx.x * 256 + threadIdx.x;
  if (i < NN * HH) layer1[i] = p[i] + p[(size_t)NN * HH + i];
}

// ------------------------------- gather rows + build bf16 copies (+ transposes)
__global__ void k_gather(const float* __restrict__ layer1,
                         const int* __restrict__ qidx, const int* __restrict__ aidx,
                         unsigned short* __restrict__ Qbf, unsigned short* __restrict__ QbfT,
                         unsigned short* __restrict__ Abf, unsigned short* __restrict__ AbfT) {
  int i = blockIdx.x * 256 + threadIdx.x;
  if (i >= BB * LL * HH) return;
  int h = i & (HH - 1);
  int l = (i >> 8) & (LL - 1);
  int b = i >> 15;
  int qi = qidx[b * LL + l];
  int ai = aidx[b * LL + l];
  unsigned short qv = f2bf(layer1[(size_t)qi * HH + h]);
  unsigned short av = f2bf(layer1[(size_t)ai * HH + h]);
  Qbf[i] = qv;
  QbfT[((size_t)b * HH + h) * LL + l] = qv;
  Abf[i] = av;
  AbfT[((size_t)b * HH + h) * LL + l] = av;
}

// ------------------------------------------------ score[b] = Q @ A^T (K = 256)
__global__ __launch_bounds__(256) void k_score(const unsigned short* __restrict__ Qbf,
                                               const unsigned short* __restrict__ Abf,
                                               float* __restrict__ score) {
  const int b = blockIdx.x;
  const int tid = threadIdx.x, wave = tid >> 5, lane = tid & 31;
  const int lrow = lane & 15, hi = (lane >> 4) & 1;
  const unsigned short* Qb = Qbf + (size_t)b * LL * HH;
  const unsigned short* Ab = Abf + (size_t)b * LL * HH;
  const int tm = wave;
  v8f acc[8] = {};
  for (int kk = 0; kk < HH; kk += 32) {
    ABf a;
    const unsigned short* ap = Qb + (size_t)(tm * 16 + lrow) * HH + kk + hi * 16;
    a.q[0] = *(const uint4*)(ap);
    a.q[1] = *(const uint4*)(ap + 8);
    const int kb = kk + (hi << 4);
    ABf bb[8];
#pragma unroll
    for (int tn = 0; tn < 8; ++tn) {
      const uint4* bp = (const uint4*)(Ab + (size_t)(tn * 16 + lrow) * HH + kb);
      bb[tn].q[0] = bp[0];
      bb[tn].q[1] = bp[1];
    }
    SCHED_FENCE();
#pragma unroll
    for (int tn = 0; tn < 8; ++tn)
      acc[tn] = wmma_bf16(a, bb[tn], acc[tn]);
  }
  float* Sb = score + (size_t)b * LL * LL;
#pragma unroll
  for (int tn = 0; tn < 8; ++tn)
#pragma unroll
    for (int r = 0; r < 8; ++r)
      Sb[(size_t)(tm * 16 + r + hi * 8) * LL + tn * 16 + lrow] = acc[tn][r];
}

// ------------------------ wq = softmax(rows), waT[m][l] = softmax(cols)(l,m)
__global__ __launch_bounds__(256) void k_softmax(const float* __restrict__ score,
                                                 unsigned short* __restrict__ wq,
                                                 unsigned short* __restrict__ waT) {
  const int b = blockIdx.x;
  const float* S = score + (size_t)b * LL * LL;
  __shared__ float rmax[LL], rinv[LL], cmax[LL], cinv[LL];
  const int t = threadIdx.x;
  if (t < LL) {
    float m = -3.4e38f;
    for (int j = 0; j < LL; ++j) m = fmaxf(m, S[t * LL + j]);
    float s = 0.f;
    for (int j = 0; j < LL; ++j) s += __expf(S[t * LL + j] - m);
    rmax[t] = m;
    rinv[t] = 1.f / s;
  } else {
    int c = t - LL;
    float m = -3.4e38f;
    for (int i = 0; i < LL; ++i) m = fmaxf(m, S[i * LL + c]);
    float s = 0.f;
    for (int i = 0; i < LL; ++i) s += __expf(S[i * LL + c] - m);
    cmax[c] = m;
    cinv[c] = 1.f / s;
  }
  __syncthreads();
  unsigned short* wqb = wq + (size_t)b * LL * LL;
  unsigned short* wab = waT + (size_t)b * LL * LL;
  for (int i = t; i < LL * LL; i += 256) {
    int l = i >> 7, m = i & (LL - 1);
    float s = S[i];
    wqb[i] = f2bf(__expf(s - rmax[l]) * rinv[l]);
    wab[m * LL + l] = f2bf(__expf(s - cmax[m]) * cinv[m]);
  }
}

// ------------- E = W(128x128) @ V (via Vt[256][128]); fused match features.
// 512 threads: wave pair shares M tile row; each wave does 8 of 16 N tiles.
// tpad[b][5+m][5+h]    = (src - e)^2
// tpad[b][5+m][5+HH+h] = src * e        (bf16, zero-padded borders)
__global__ __launch_bounds__(512) void k_align(const unsigned short* __restrict__ W,
                                               const unsigned short* __restrict__ Vt,
                                               const unsigned short* __restrict__ Src,
                                               unsigned short* __restrict__ tpad) {
  const int b = blockIdx.x;
  const int tid = threadIdx.x, wave = tid >> 5, lane = tid & 31;
  const int lrow = lane & 15, hi = (lane >> 4) & 1;
  const int wrow = wave >> 1;     // M tile 0..7
  const int wcol = wave & 1;      // N half
  const unsigned short* Wb = W + (size_t)b * LL * LL;
  const unsigned short* Vb = Vt + (size_t)b * HH * LL;
  const unsigned short* Sb = Src + (size_t)b * LL * HH;
  unsigned short* Tp = tpad + (size_t)b * PADH * PADW;
  v8f acc[8] = {};
  for (int kk = 0; kk < LL; kk += 32) {
    ABf a;
    const unsigned short* ap = Wb + (size_t)(wrow * 16 + lrow) * LL + kk + hi * 16;
    a.q[0] = *(const uint4*)(ap);
    a.q[1] = *(const uint4*)(ap + 8);
    const int kb = kk + (hi << 4);
    ABf bb[8];
#pragma unroll
    for (int tn = 0; tn < 8; ++tn) {
      const uint4* bp = (const uint4*)(Vb + (size_t)((wcol * 8 + tn) * 16 + lrow) * LL + kb);
      bb[tn].q[0] = bp[0];
      bb[tn].q[1] = bp[1];
    }
    SCHED_FENCE();
#pragma unroll
    for (int tn = 0; tn < 8; ++tn)
      acc[tn] = wmma_bf16(a, bb[tn], acc[tn]);
  }
#pragma unroll
  for (int tn = 0; tn < 8; ++tn)
#pragma unroll
    for (int r = 0; r < 8; ++r) {
      int m = wrow * 16 + r + hi * 8;
      int h = (wcol * 8 + tn) * 16 + lrow;
      float e = acc[tn][r];
      float s = bf2f(Sb[(size_t)m * HH + h]);
      float d = (s - e) * (s - e);
      float p = s * e;
      Tp[(5 + m) * PADW + 5 + h]      = f2bf(d);
      Tp[(5 + m) * PADW + 5 + HH + h] = f2bf(p);
    }
}

// ------------------------ conv weights -> bf16, packed [112][KP], zero padded
__global__ void k_convw(const float* __restrict__ w0, const float* __restrict__ w1,
                        const float* __restrict__ w2, const float* __restrict__ w3,
                        unsigned short* __restrict__ wbf) {
  const int KPt[4] = {544, 1056, 1568, 2112};
  const int OFF[4] = {0, 60928, 179200, 354816};
  int f = blockIdx.y;
  int KP = KPt[f];
  int i = blockIdx.x * 256 + threadIdx.x;
  if (i >= 112 * KP) return;
  int ko = i / KP;
  int k = i - ko * KP;
  int kreal = 522 * (f + 1);
  const float* ws_ = (f == 0) ? w0 : (f == 1) ? w1 : (f == 2) ? w2 : w3;
  unsigned short v = 0;
  if (ko < 100 && k < kreal) v = f2bf(ws_[(size_t)ko * kreal + k]);
  wbf[OFF[f] + i] = v;
}

// ------------- conv as GEMM: M = output rows (oh), N = 100 filters (pad 112),
// K = fs*522 (pad to mult of 32). A-operand gathered from padded feature map.
__global__ __launch_bounds__(256) void k_conv(const unsigned short* __restrict__ tq,
                                              const unsigned short* __restrict__ ta,
                                              const unsigned short* __restrict__ wbf,
                                              const float* __restrict__ b0,
                                              const float* __restrict__ b1,
                                              const float* __restrict__ b2,
                                              const float* __restrict__ b3,
                                              float* __restrict__ ybuf) {
  const int KPt[4] = {544, 1056, 1568, 2112};
  const int OFF[4] = {0, 60928, 179200, 354816};
  const int f = blockIdx.y & 3;
  const int side = blockIdx.y >> 2;
  const int b = blockIdx.x;
  const int KP = KPt[f];
  const int kreal = 522 * (f + 1);
  const int OH = 138 - f;
  const unsigned short* Tp = (side ? ta : tq) + (size_t)b * PADH * PADW;
  const unsigned short* Wf = wbf + OFF[f];
  const float* bias = (f == 0) ? b0 : (f == 1) ? b1 : (f == 2) ? b2 : b3;
  float* yb = ybuf + ((size_t)(side * 4 + f) * BB + b) * 144 * 112;

  const int tid = threadIdx.x, wave = tid >> 5, lane = tid & 31;
  const int lrow = lane & 15, hi = (lane >> 4) & 1;

  for (int tm = wave; tm < 9; tm += 8) {
    v8f acc[7] = {};
    const int ohA = tm * 16 + lrow;
    for (int kk = 0; kk < KP; kk += 32) {
      const int kb = kk + (hi << 4);
      ABf bb[7];
#pragma unroll
      for (int tn = 0; tn < 7; ++tn) {
        const uint4* bp = (const uint4*)(Wf + (size_t)(tn * 16 + lrow) * KP + kb);
        bb[tn].q[0] = bp[0];
        bb[tn].q[1] = bp[1];
      }
      ABf a;
#pragma unroll
      for (int ch = 0; ch < 2; ++ch) {
        const int kb0 = kk + hi * 16 + ch * 8;
#pragma unroll
        for (int e = 0; e < 8; ++e) {
          const int k = kb0 + e;
          unsigned short v = 0;
          if (k < kreal) {
            const int r = k / 522;
            const int c = k - r * 522;
            v = Tp[(ohA + r) * PADW + c];
          }
          a.s[ch * 8 + e] = v;
        }
      }
      SCHED_FENCE();
#pragma unroll
      for (int tn = 0; tn < 7; ++tn)
        acc[tn] = wmma_bf16(a, bb[tn], acc[tn]);
    }
#pragma unroll
    for (int tn = 0; tn < 7; ++tn)
#pragma unroll
      for (int r = 0; r < 8; ++r) {
        const int oh = tm * 16 + r + hi * 8;
        const int ko = tn * 16 + lrow;
        if (oh < OH && ko < 100) {
          float y = acc[tn][r] + bias[ko];
          yb[(size_t)oh * 112 + ko] = fmaxf(y, 0.f);
        }
      }
  }
}

// --------------------------------------- max over oh -> feat[B][800] (relu'd)
__global__ void k_maxred(const float* __restrict__ ybuf, float* __restrict__ feat) {
  int i = blockIdx.x * 256 + threadIdx.x;
  if (i >= BB * 800) return;
  int b = i / 800;
  int j = i - b * 800;
  int side = j / 400;
  int jj = j - side * 400;
  int f = jj / 100;
  int k = jj - f * 100;
  int OH = 138 - f;
  const float* yb = ybuf + ((size_t)(side * 4 + f) * BB + b) * 144 * 112;
  float m = 0.f;   // values are post-relu, so 0 is the correct identity
  for (int oh = 0; oh < OH; ++oh) m = fmaxf(m, yb[(size_t)oh * 112 + k]);
  feat[(size_t)b * 800 + j] = m;
}

// ------------------------------------------------- dense (800->2) + log_softmax
__global__ void k_dense(const float* __restrict__ feat, const float* __restrict__ dw,
                        const float* __restrict__ db, float* __restrict__ out) {
  int b = threadIdx.x;
  if (b >= BB) return;
  float l0 = db[0], l1 = db[1];
  const float* fb = feat + (size_t)b * 800;
  for (int j = 0; j < 800; ++j) {
    float v = fb[j];
    l0 += v * dw[2 * j];
    l1 += v * dw[2 * j + 1];
  }
  float m = fmaxf(l0, l1);
  float lse = m + __logf(__expf(l0 - m) + __expf(l1 - m));
  out[2 * b]     = l0 - lse;
  out[2 * b + 1] = l1 - lse;
}

extern "C" void kernel_launch(void* const* d_in, const int* in_sizes, int n_in,
                              void* d_out, int out_size, void* d_ws, size_t ws_size,
                              hipStream_t stream) {
  const float* adj  = (const float*)d_in[0];
  const float* emb  = (const float*)d_in[1];
  const int*   qidx = (const int*)d_in[2];
  const int*   aidx = (const int*)d_in[3];
  const float* cw0  = (const float*)d_in[4];
  const float* cb0  = (const float*)d_in[5];
  const float* cw1  = (const float*)d_in[6];
  const float* cb1  = (const float*)d_in[7];
  const float* cw2  = (const float*)d_in[8];
  const float* cb2  = (const float*)d_in[9];
  const float* cw3  = (const float*)d_in[10];
  const float* cb3  = (const float*)d_in[11];
  const float* dw   = (const float*)d_in[12];
  const float* db   = (const float*)d_in[13];
  float* out = (float*)d_out;
  (void)in_sizes; (void)n_in; (void)out_size; (void)ws_size;

  char* base = (char*)d_ws;
  size_t off = 0;
  auto take = [&](size_t bytes) -> void* {
    void* p = base + off;
    off = (off + bytes + 255) & ~(size_t)255;
    return p;
  };
  unsigned short* embT = (unsigned short*)take((size_t)HH * NN * 2);
  float* part          = (float*)take((size_t)2 * NN * HH * 4);
  float* layer1        = (float*)take((size_t)NN * HH * 4);
  unsigned short* Qbf  = (unsigned short*)take((size_t)BB * LL * HH * 2);
  unsigned short* QbfT = (unsigned short*)take((size_t)BB * HH * LL * 2);
  unsigned short* Abf  = (unsigned short*)take((size_t)BB * LL * HH * 2);
  unsigned short* AbfT = (unsigned short*)take((size_t)BB * HH * LL * 2);
  float* score         = (float*)take((size_t)BB * LL * LL * 4);
  unsigned short* wq   = (unsigned short*)take((size_t)BB * LL * LL * 2);
  unsigned short* waT  = (unsigned short*)take((size_t)BB * LL * LL * 2);
  unsigned short* tq   = (unsigned short*)take((size_t)BB * PADH * PADW * 2);
  unsigned short* ta   = (unsigned short*)take((size_t)BB * PADH * PADW * 2);
  unsigned short* wbf  = (unsigned short*)take((size_t)591360 * 2);
  float* ybuf          = (float*)take((size_t)8 * BB * 144 * 112 * 4);
  float* feat          = (float*)take((size_t)BB * 800 * 4);

  k_embT<<<(NN * HH + 255) / 256, 256, 0, stream>>>(emb, embT);
  k_gemm1<<<dim3(64, 2), 512, 0, stream>>>(adj, embT, part);
  k_add<<<(NN * HH + 255) / 256, 256, 0, stream>>>(part, layer1);
  k_gather<<<(BB * LL * HH + 255) / 256, 256, 0, stream>>>(layer1, qidx, aidx,
                                                           Qbf, QbfT, Abf, AbfT);
  k_score<<<BB, 256, 0, stream>>>(Qbf, Abf, score);
  k_softmax<<<BB, 256, 0, stream>>>(score, wq, waT);
  k_zero<<<2048, 256, 0, stream>>>((unsigned int*)tq,
                                   (long long)((size_t)BB * PADH * PADW * 2 / 4));
  k_zero<<<2048, 256, 0, stream>>>((unsigned int*)ta,
                                   (long long)((size_t)BB * PADH * PADW * 2 / 4));
  k_align<<<BB, 512, 0, stream>>>(wq, AbfT, Qbf, tq);
  k_align<<<BB, 512, 0, stream>>>(waT, QbfT, Abf, ta);
  k_convw<<<dim3((112 * 2112 + 255) / 256, 4), 256, 0, stream>>>(cw0, cw1, cw2, cw3, wbf);
  k_conv<<<dim3(BB, 8), 256, 0, stream>>>(tq, ta, wbf, cb0, cb1, cb2, cb3, ybuf);
  k_maxred<<<(BB * 800 + 255) / 256, 256, 0, stream>>>(ybuf, feat);
  k_dense<<<1, 32, 0, stream>>>(feat, dw, db, out);
}
